// LengthRegulator_13692355739900
// MI455X (gfx1250) — matrix operand order repaired and checked by reading, not compiled, not run on hardware
//
#include <hip/hip_runtime.h>

typedef __bf16 bf16_t;
typedef __attribute__((ext_vector_type(16))) __bf16 v16bf;
typedef __attribute__((ext_vector_type(8)))  float  v8f;

#define NB    32          // batch
#define SEQ   1024        // sequence length
#define CH    256         // channels (D == F == 256)
#define TMAX  4096        // mel_max_length (fixed by reference setup)
#define KTOT  768         // 3 * 256 reduction length of each conv-GEMM
#define MTILE 64          // output rows per block (2 m-tiles per wave)
#define XROWS 66          // MTILE + 2 halo rows
#define XSTR  264         // LDS row stride in bf16 elems (256 + 8 pad)

// ---- workspace layout (bytes) ----
#define WS_W1T 0
#define WS_W2T (KTOT * CH * 2)
#define WS_H1  (2 * KTOT * CH * 2)
#define WS_CUM (WS_H1 + (size_t)NB * SEQ * CH * 2)

// ============================================================
// Repack conv weights [K][in][out] fp32 -> [out][K*in] bf16
// ============================================================
__global__ void prep_weights(const float* __restrict__ W1,
                             const float* __restrict__ W2,
                             bf16_t* __restrict__ W1T,
                             bf16_t* __restrict__ W2T) {
  int idx = blockIdx.x * blockDim.x + threadIdx.x;
  if (idx >= 2 * KTOT * CH) return;
  int which = idx >= KTOT * CH;
  int rem   = which ? idx - KTOT * CH : idx;
  int n = rem / KTOT;
  int k = rem - n * KTOT;
  int tap = k >> 8;
  int c   = k & 255;
  const float* Wsrc = which ? W2 : W1;
  bf16_t*      Wdst = which ? W2T : W1T;
  Wdst[(size_t)n * KTOT + k] = (bf16_t)Wsrc[((size_t)tap * CH + c) * CH + n];
}

// ============================================================
// Conv1D(K=3) as WMMA GEMM + bias + LayerNorm + ReLU.
// Block tile: 64 rows x 256 cols. 8 waves; each wave owns
// 2 m-tiles x 4 n-tiles (8 accumulators) so every B fragment
// is reused across two A fragments (halves L2 weight traffic).
// LAYER==1: fp32 input * mask, writes bf16 H1.
// LAYER==2: bf16 H1 input, fuses linear(256->1)+mask+ReLU -> dur.
// ============================================================
template <int LAYER>
__global__ __launch_bounds__(256) void dp_conv_kernel(
    const float*  __restrict__ x,      // [NB*SEQ, CH] fp32 (layer 1)
    const bf16_t* __restrict__ h_in,   // [NB*SEQ, CH] bf16 (layer 2)
    const float*  __restrict__ mask,   // [NB*SEQ]
    const bf16_t* __restrict__ WT,     // [CH][KTOT] bf16
    const float*  __restrict__ bias,
    const float*  __restrict__ gamma,
    const float*  __restrict__ beta,
    const float*  __restrict__ Wl,     // [CH] (layer 2)
    const float*  __restrict__ bl,     // [1]  (layer 2)
    bf16_t* __restrict__ h_out,        // [NB*SEQ, CH] (layer 1)
    float*  __restrict__ dur_out) {    // [NB*SEQ]     (layer 2)
  __shared__ bf16_t xs[XROWS * XSTR];
  __shared__ float  ctile[MTILE][CH];

  const int tid = threadIdx.x;
  const int blk = blockIdx.x;
  const int b   = blk >> 4;            // 16 blocks per batch row
  const int s0  = (blk & 15) * MTILE;
  const int m0  = b * SEQ + s0;

  // ---- stage input rows s0-1 .. s0+64 into LDS as bf16 ----
  if (LAYER == 1) {
    for (int i = tid; i < XROWS * 64; i += 256) {
      int row = i >> 6;
      int c4  = (i & 63) * 4;
      int s   = s0 - 1 + row;
      float4 v = make_float4(0.f, 0.f, 0.f, 0.f);
      if (s >= 0 && s < SEQ) {
        v = *(const float4*)(x + ((size_t)b * SEQ + s) * CH + c4);
        float mv = mask[b * SEQ + s];
        v.x *= mv; v.y *= mv; v.z *= mv; v.w *= mv;
      }
      union { bf16_t h[4]; uint2 u; } pk;
      pk.h[0] = (bf16_t)v.x; pk.h[1] = (bf16_t)v.y;
      pk.h[2] = (bf16_t)v.z; pk.h[3] = (bf16_t)v.w;
      *(uint2*)(xs + row * XSTR + c4) = pk.u;
    }
  } else {
    for (int i = tid; i < XROWS * 32; i += 256) {
      int row = i >> 5;
      int c8  = (i & 31) * 8;
      int s   = s0 - 1 + row;
      uint4 v = make_uint4(0u, 0u, 0u, 0u);
      if (s >= 0 && s < SEQ)
        v = *(const uint4*)(h_in + ((size_t)b * SEQ + s) * CH + c8);
      *(uint4*)(xs + row * XSTR + c8) = v;
    }
  }
  __syncthreads();

  // ---- WMMA main loop: C[64x256] += A[64x768] * B[768x256] ----
  const int wave = tid >> 5;
  const int lane = tid & 31;
  const int lo16 = (lane < 16) ? 1 : 0;
  const int l15  = lane & 15;
  const int mg   = wave >> 2;           // 0..1 : which 32-row group
  const int ntb  = (wave & 3) * 64;     // wave's 64-column slab

  v8f acc[2][4] = {};

  const int arow = mg * 32 + l15;       // A lane row base (add mi*16 + tap)
  const int akb  = lo16 ? 0 : 8;        // A lane K base (ISA 16b A layout)
  const int bko  = lo16 ? 0 : 8;        // B lane K base (ISA 16b B layout)

  for (int kk = 0; kk < 24; ++kk) {
    const int tap = kk >> 3;            // 8 k-chunks per tap (256/32)
    const int c0  = (kk & 7) * 32;

    // two A fragments (m-tiles at rows +0 and +16)
    union { uint4 q[2]; v16bf v; } afr[2];
#pragma unroll
    for (int mi = 0; mi < 2; ++mi) {
      const bf16_t* ap = xs + (arow + mi * 16 + tap) * XSTR + c0 + akb;
      afr[mi].q[0] = *(const uint4*)(ap);       // K = kb .. kb+7
      afr[mi].q[1] = *(const uint4*)(ap + 16);  // K = kb+16 .. kb+23
    }

    const bf16_t* wbase = WT + ((size_t)(ntb + l15)) * KTOT + kk * 32 + bko;
    __builtin_prefetch(wbase + 32, 0, 1);   // next k-chunk -> global_prefetch_b8
#pragma unroll
    for (int t = 0; t < 4; ++t) {
      union { uint4 q[2]; v16bf v; } bfr;
      const bf16_t* bp = wbase + (size_t)t * 16 * KTOT;
      bfr.q[0] = *(const uint4*)(bp);
      bfr.q[1] = *(const uint4*)(bp + 16);
      // reuse B across both m-tiles
      acc[0][t] = __builtin_amdgcn_wmma_f32_16x16x32_bf16(
          false, afr[0].v, false, bfr.v, (short)0, acc[0][t], false, false);
      acc[1][t] = __builtin_amdgcn_wmma_f32_16x16x32_bf16(
          false, afr[1].v, false, bfr.v, (short)0, acc[1][t], false, false);
    }
  }

  // ---- epilogue: accumulators (+bias) -> LDS fp32 tile ----
#pragma unroll
  for (int t = 0; t < 4; ++t) {
    int ncol = ntb + t * 16 + l15;
    float bv = bias[ncol];
#pragma unroll
    for (int mi = 0; mi < 2; ++mi) {
      int rowbase = mg * 32 + mi * 16 + (lo16 ? 0 : 8);
#pragma unroll
      for (int r = 0; r < 8; ++r)
        ctile[rowbase + r][ncol] = acc[mi][t][r] + bv;
    }
  }
  __syncthreads();

  // ---- LayerNorm + ReLU (+ fused linear for layer 2), wave per 8 rows ----
  for (int rr = 0; rr < 8; ++rr) {
    int row = wave * 8 + rr;
    float vals[8];
    float s = 0.f, ss = 0.f;
#pragma unroll
    for (int j = 0; j < 8; ++j) {
      float v = ctile[row][lane * 8 + j];
      vals[j] = v; s += v; ss += v * v;
    }
#pragma unroll
    for (int o = 16; o > 0; o >>= 1) {
      s  += __shfl_xor(s,  o, 32);
      ss += __shfl_xor(ss, o, 32);
    }
    float mu   = s * (1.f / 256.f);
    float var  = ss * (1.f / 256.f) - mu * mu;
    float rstd = rsqrtf(var + 1e-5f);
    int m = m0 + row;
    if (LAYER == 1) {
      union { bf16_t h[8]; uint4 u; } pk;
#pragma unroll
      for (int j = 0; j < 8; ++j) {
        int c = lane * 8 + j;
        float y = (vals[j] - mu) * rstd * gamma[c] + beta[c];
        pk.h[j] = (bf16_t)(y > 0.f ? y : 0.f);
      }
      *(uint4*)(h_out + (size_t)m * CH + lane * 8) = pk.u;
    } else {
      float dot = 0.f;
#pragma unroll
      for (int j = 0; j < 8; ++j) {
        int c = lane * 8 + j;
        float y = (vals[j] - mu) * rstd * gamma[c] + beta[c];
        y = y > 0.f ? y : 0.f;
        dot += y * Wl[c];
      }
#pragma unroll
      for (int o = 16; o > 0; o >>= 1) dot += __shfl_xor(dot, o, 32);
      if (lane == 0) {
        float dv = (dot + bl[0]) * mask[m];
        dur_out[m] = dv > 0.f ? dv : 0.f;
      }
    }
  }
}

// ============================================================
// Per-batch inclusive cumsum of reps = round_half_up(target)+1
// ============================================================
__global__ __launch_bounds__(256) void scan_kernel(const int* __restrict__ target,
                                                   int* __restrict__ cum) {
  __shared__ int tsum[256];
  int b = blockIdx.x, tid = threadIdx.x;
  int local[4];
  int s = 0;
#pragma unroll
  for (int j = 0; j < 4; ++j) {
    int tgt = target[b * SEQ + tid * 4 + j];
    int rep = (int)floorf((float)tgt * 1.0f + 0.5f) + 1;   // ALPHA = 1.0
    s += rep;
    local[j] = s;
  }
  tsum[tid] = s;
  __syncthreads();
  for (int off = 1; off < 256; off <<= 1) {
    int v   = tsum[tid];
    int add = (tid >= off) ? tsum[tid - off] : 0;
    __syncthreads();
    tsum[tid] = v + add;
    __syncthreads();
  }
  int excl = (tid == 0) ? 0 : tsum[tid - 1];
#pragma unroll
  for (int j = 0; j < 4; ++j) cum[b * SEQ + tid * 4 + j] = excl + local[j];
}

// ============================================================
// Length regulation: searchsorted + row gather (+ decoder_pos)
// 4 output rows per block; 64 lanes x float4 per row.
// ============================================================
__global__ __launch_bounds__(256) void gather_kernel(const float* __restrict__ enc,
                                                     const int* __restrict__ cum,
                                                     float* __restrict__ out,
                                                     float* __restrict__ pos) {
  __shared__ int cumS[SEQ];
  int b     = blockIdx.x >> 10;          // 1024 blocks per batch
  int tbase = (blockIdx.x & 1023) * 4;
  int tid   = threadIdx.x;
  for (int i = tid; i < SEQ; i += 256) cumS[i] = cum[b * SEQ + i];
  __syncthreads();

  int rsub = tid >> 6;
  int c4   = (tid & 63) * 4;
  int t    = tbase + rsub;
  int total = cumS[SEQ - 1];

  int p = 0;                              // searchsorted(cum, t, 'right')
#pragma unroll
  for (int step = 512; step > 0; step >>= 1)
    if (p + step <= SEQ && cumS[p + step - 1] <= t) p += step;
  int valid = (t < total) ? 1 : 0;
  int idx   = p < (SEQ - 1) ? p : (SEQ - 1);

  float4 v = make_float4(0.f, 0.f, 0.f, 0.f);
  if (valid) v = *(const float4*)(enc + ((size_t)b * SEQ + idx) * CH + c4);
  *(float4*)(out + ((size_t)b * TMAX + t) * CH + c4) = v;
  if (c4 == 0) pos[(size_t)b * TMAX + t] = valid ? (float)(t + 1) : 0.f;
}

// ============================================================
extern "C" void kernel_launch(void* const* d_in, const int* in_sizes, int n_in,
                              void* d_out, int out_size, void* d_ws, size_t ws_size,
                              hipStream_t stream) {
  const float* enc    = (const float*)d_in[0];
  const float* mask   = (const float*)d_in[1];
  const int*   target = (const int*)d_in[2];
  // d_in[3] = mel_max_length (device scalar; reference fixes it to 4096)
  const float* W1  = (const float*)d_in[4];
  const float* b1  = (const float*)d_in[5];
  const float* g1  = (const float*)d_in[6];
  const float* be1 = (const float*)d_in[7];
  const float* W2  = (const float*)d_in[8];
  const float* b2  = (const float*)d_in[9];
  const float* g2  = (const float*)d_in[10];
  const float* be2 = (const float*)d_in[11];
  const float* Wl  = (const float*)d_in[12];
  const float* bl  = (const float*)d_in[13];

  char*   ws  = (char*)d_ws;
  bf16_t* W1T = (bf16_t*)(ws + WS_W1T);
  bf16_t* W2T = (bf16_t*)(ws + WS_W2T);
  bf16_t* H1  = (bf16_t*)(ws + WS_H1);
  int*    cum = (int*)(ws + WS_CUM);

  float* out_main = (float*)d_out;                               // [32,4096,256]
  float* out_pos  = out_main + (size_t)NB * TMAX * CH;           // [32,4096]
  float* out_dur  = out_pos + (size_t)NB * TMAX;                 // [32,1024]

  prep_weights<<<(2 * KTOT * CH + 255) / 256, 256, 0, stream>>>(W1, W2, W1T, W2T);
  scan_kernel<<<NB, 256, 0, stream>>>(target, cum);
  dp_conv_kernel<1><<<(NB * SEQ) / MTILE, 256, 0, stream>>>(
      enc, nullptr, mask, W1T, b1, g1, be1, nullptr, nullptr, H1, nullptr);
  dp_conv_kernel<2><<<(NB * SEQ) / MTILE, 256, 0, stream>>>(
      nullptr, H1, mask, W2T, b2, g2, be2, Wl, bl, nullptr, out_dur);
  gather_kernel<<<NB * (TMAX / 4), 256, 0, stream>>>(enc, cum, out_main, out_pos);
}
